// GATConv_67508295958889
// MI455X (gfx1250) — matrix operand order repaired and checked by reading, not compiled, not run on hardware
//
#include <hip/hip_runtime.h>
#include <hip/hip_bf16.h>

#define NUM_HEADS 4
#define OUT_FEATS 64
#define IN_FEATS 256
#define HF (NUM_HEADS * OUT_FEATS)   // 256
#define NEG_SLOPE 0.2f

typedef __attribute__((ext_vector_type(16))) _Float16 v16h;
typedef __attribute__((ext_vector_type(8)))  _Float16 v8h;
typedef __attribute__((ext_vector_type(8)))  float    v8f;

// ---------------------------------------------------------------------------
// Kernel 0a: convert feat f32 -> f16 (grid-stride)
// ---------------------------------------------------------------------------
__global__ void gat_convert_feat(const float* __restrict__ feat,
                                 _Float16* __restrict__ feat16, long long nelem) {
    long long i = (long long)blockIdx.x * blockDim.x + threadIdx.x;
    long long stride = (long long)gridDim.x * blockDim.x;
    for (; i < nelem; i += stride) feat16[i] = (_Float16)feat[i];
}

// ---------------------------------------------------------------------------
// Kernel 0b: convert + transpose W [IN_FEATS, HF] f32 -> Wt [HF, IN_FEATS] f16
// ---------------------------------------------------------------------------
__global__ void gat_convert_Wt(const float* __restrict__ W,
                               _Float16* __restrict__ Wt) {
    int idx = blockIdx.x * blockDim.x + threadIdx.x;   // over HF*IN_FEATS
    if (idx >= HF * IN_FEATS) return;
    int n = idx / IN_FEATS;       // output column
    int k = idx - n * IN_FEATS;   // input feature
    Wt[(size_t)n * IN_FEATS + k] = (_Float16)W[(size_t)k * HF + n];
}

// ---------------------------------------------------------------------------
// Kernel 1: h = feat @ W via V_WMMA_F32_16X16X32_F16
// 128 threads = 4 waves per block; block owns 16 rows; wave w owns head w
// (cols [64w, 64w+64) as 4 column tiles). K = 256 -> 8 WMMA steps per tile.
// A fragment (16x32 f16, ISA 7.12.2): lane m = lane&15; lane<16 holds
// K = {k0..k0+7, k0+16..k0+23}, lane>=16 holds K = {k0+8..k0+15, k0+24..k0+31}.
// B fragment (32x16 f16): lane n = lane&15; lanes 0-15 hold K={k0..k0+7, k0+16..k0+23},
// lanes 16-31 hold K={k0+8..k0+15, k0+24..k0+31} (same order as A; 8-runs contiguous in Wt [n][k]).
// ---------------------------------------------------------------------------
__global__ __launch_bounds__(128)
void gat_gemm_wmma(const _Float16* __restrict__ feat16,
                   const _Float16* __restrict__ Wt,
                   _Float16* __restrict__ h16, int n_nodes) {
    const int wave = threadIdx.x >> 5;
    const int lane = threadIdx.x & 31;
    const int r    = lane & 15;
    const int hi   = lane >> 4;
    const int m0   = blockIdx.x * 16;
    const int ncol_base = wave * 64;

    int row_a = m0 + r;
    if (row_a > n_nodes - 1) row_a = n_nodes - 1;   // clamp loads; EXEC stays all-1s
    const _Float16* arow = feat16 + (size_t)row_a * IN_FEATS;

    v8f acc[4] = {v8f{}, v8f{}, v8f{}, v8f{}};

    for (int k0 = 0; k0 < IN_FEATS; k0 += 32) {
        if (k0 + 64 < IN_FEATS)
            __builtin_prefetch(arow + k0 + 64, 0, 1);   // global_prefetch_b8

        v8h alo = *(const v8h*)(arow + k0 + hi * 8);
        v8h ahi = *(const v8h*)(arow + k0 + hi * 8 + 16);
        v16h a;
#pragma unroll
        for (int i = 0; i < 8; ++i) { a[i] = alo[i]; a[i + 8] = ahi[i]; }

#pragma unroll
        for (int t = 0; t < 4; ++t) {
            const _Float16* bcol =
                Wt + (size_t)(ncol_base + t * 16 + r) * IN_FEATS + k0 + hi * 8;
            v8h blo = *(const v8h*)(bcol);
            v8h bhi = *(const v8h*)(bcol + 16);
            v16h b;
#pragma unroll
            for (int i = 0; i < 8; ++i) { b[i] = blo[i]; b[i + 8] = bhi[i]; }
            acc[t] = __builtin_amdgcn_wmma_f32_16x16x32_f16(
                false, a, false, b, (short)0, acc[t], false, false);
        }
    }

    // D layout: VGPR v, lane -> (m = v + 8*hi, n = lane&15)
#pragma unroll
    for (int t = 0; t < 4; ++t) {
#pragma unroll
        for (int v = 0; v < 8; ++v) {
            int row = m0 + v + 8 * hi;
            if (row < n_nodes) {
                int col = ncol_base + t * 16 + r;
                h16[(size_t)row * HF + col] = (_Float16)acc[t][v];
            }
        }
    }
}

// ---------------------------------------------------------------------------
// Kernel 2: per-(node, head) attention dot products
// attn_row[n,h] = sum_f attn_l[h,f]*h[n,h,f]; attn_col likewise with attn_r
// ---------------------------------------------------------------------------
__global__ void gat_attn_scores(const _Float16* __restrict__ h16,
                                const float* __restrict__ attn_l,
                                const float* __restrict__ attn_r,
                                float* __restrict__ attn_row,
                                float* __restrict__ attn_col, int n_nodes) {
    int t = blockIdx.x * blockDim.x + threadIdx.x;
    if (t >= n_nodes * NUM_HEADS) return;
    int node = t >> 2;
    int hd   = t & 3;
    const _Float16* hp = h16 + (size_t)node * HF + hd * OUT_FEATS;
    const float* al = attn_l + hd * OUT_FEATS;
    const float* ar = attn_r + hd * OUT_FEATS;
    float sl = 0.f, sr = 0.f;
#pragma unroll 8
    for (int f = 0; f < OUT_FEATS; ++f) {
        float v = (float)hp[f];
        sl += al[f] * v;
        sr += ar[f] * v;
    }
    attn_row[t] = sl;
    attn_col[t] = sr;
}

// ---------------------------------------------------------------------------
// Kernel 3: wave-per-destination segment softmax + gather aggregation.
// Lane l owns output elements [8l, 8l+8) of the 256-wide row -> head = l>>3
// is lane-constant; per-neighbor scalar loads are wave-broadcasts; feature
// gathers are 16B/lane (512B/wave, coalesced, hitting L2-resident h16).
// ---------------------------------------------------------------------------
__global__ __launch_bounds__(256)
void gat_softmax_aggregate(const int* __restrict__ row_ptr,
                           const int* __restrict__ col_ind,
                           const _Float16* __restrict__ h16,
                           const float* __restrict__ attn_row,
                           const float* __restrict__ attn_col,
                           float* __restrict__ out, int n_nodes) {
    int wave = threadIdx.x >> 5;
    int lane = threadIdx.x & 31;
    int node = blockIdx.x * 8 + wave;
    if (node >= n_nodes) return;

    int e0 = row_ptr[node];
    int e1 = row_ptr[node + 1];
    int hd = lane >> 3;
    float ar = attn_row[node * NUM_HEADS + hd];

    // pass 1: max
    float m = -3.4e38f;
    for (int e = e0; e < e1; ++e) {
        int src = col_ind[e];
        float s = ar + attn_col[src * NUM_HEADS + hd];
        s = (s >= 0.f) ? s : NEG_SLOPE * s;
        m = fmaxf(m, s);
    }
    // pass 2: sum of exp
    float sum = 0.f;
    for (int e = e0; e < e1; ++e) {
        int src = col_ind[e];
        float s = ar + attn_col[src * NUM_HEADS + hd];
        s = (s >= 0.f) ? s : NEG_SLOPE * s;
        sum += __expf(s - m);
    }
    float inv = (sum > 0.f) ? 1.f / sum : 0.f;

    // pass 3: aggregate
    float acc[8] = {0.f, 0.f, 0.f, 0.f, 0.f, 0.f, 0.f, 0.f};
    int fb = lane * 8;
    for (int e = e0; e < e1; ++e) {
        int src = col_ind[e];
        float s = ar + attn_col[src * NUM_HEADS + hd];
        s = (s >= 0.f) ? s : NEG_SLOPE * s;
        float alpha = __expf(s - m) * inv;
        v8h hv = *(const v8h*)(h16 + (size_t)src * HF + fb);
#pragma unroll
        for (int k = 0; k < 8; ++k) acc[k] += alpha * (float)hv[k];
    }
    float* op = out + (size_t)node * HF + fb;
#pragma unroll
    for (int k = 0; k < 8; ++k) op[k] = acc[k];
}

// ---------------------------------------------------------------------------
extern "C" void kernel_launch(void* const* d_in, const int* in_sizes, int n_in,
                              void* d_out, int out_size, void* d_ws, size_t ws_size,
                              hipStream_t stream) {
    const int*   row_ptr = (const int*)d_in[0];
    const int*   col_ind = (const int*)d_in[1];
    const float* feat    = (const float*)d_in[4];
    const float* W       = (const float*)d_in[5];
    const float* attn_l  = (const float*)d_in[6];
    const float* attn_r  = (const float*)d_in[7];
    float* out = (float*)d_out;

    const int n_nodes = in_sizes[0] - 1;      // row_ptr has N+1 entries
    const long long feat_elems = (long long)n_nodes * IN_FEATS;

    // carve workspace (256B-aligned segments)
    char* ws = (char*)d_ws;
    size_t off = 0;
    auto carve = [&](size_t bytes) {
        void* p = ws + off;
        off = (off + bytes + 255) & ~(size_t)255;
        return p;
    };
    _Float16* Wt16    = (_Float16*)carve((size_t)HF * IN_FEATS * sizeof(_Float16));
    _Float16* feat16  = (_Float16*)carve((size_t)feat_elems * sizeof(_Float16));
    _Float16* h16     = (_Float16*)carve((size_t)n_nodes * HF * sizeof(_Float16));
    float*    arow_ws = (float*)carve((size_t)n_nodes * NUM_HEADS * sizeof(float));
    float*    acol_ws = (float*)carve((size_t)n_nodes * NUM_HEADS * sizeof(float));
    (void)ws_size;

    // 0a: feat f32 -> f16
    {
        int blocks = (int)((feat_elems + 1023) / 1024);
        if (blocks > 4096) blocks = 4096;
        gat_convert_feat<<<blocks, 1024, 0, stream>>>(feat, feat16, feat_elems);
    }
    // 0b: W -> Wt (f16, transposed)
    {
        int elems = HF * IN_FEATS;
        gat_convert_Wt<<<(elems + 255) / 256, 256, 0, stream>>>(W, Wt16);
    }
    // 1: WMMA GEMM -> h16
    {
        int blocks = (n_nodes + 15) / 16;
        gat_gemm_wmma<<<blocks, 128, 0, stream>>>(feat16, Wt16, h16, n_nodes);
    }
    // 2: attention scores
    {
        int t = n_nodes * NUM_HEADS;
        gat_attn_scores<<<(t + 255) / 256, 256, 0, stream>>>(
            h16, attn_l, attn_r, arow_ws, acol_ws, n_nodes);
    }
    // 3: softmax + aggregate
    {
        int blocks = (n_nodes + 7) / 8;
        gat_softmax_aggregate<<<blocks, 256, 0, stream>>>(
            row_ptr, col_ind, h16, arow_ws, acol_ws, out, n_nodes);
    }
}
